// RelationNet_23081154249002
// MI455X (gfx1250) — matrix operand order repaired and checked by reading, not compiled, not run on hardware
//
#include <hip/hip_runtime.h>

typedef unsigned short u16;
typedef __attribute__((ext_vector_type(16))) __bf16 v16bf;
typedef __attribute__((ext_vector_type(8)))  float  v8f;

union Frag {
  v16bf v;
  u16   u[16];
  uint4 x[2];
};

__device__ __forceinline__ u16 f2bf(float f) {
  unsigned int x = __float_as_uint(f);
  x = (x + 0x7FFFu + ((x >> 16) & 1u)) >> 16;   // round-to-nearest-even
  return (u16)x;
}

// Generic LDS pointer -> wave-relative LDS byte offset (aperture: addr[31:0]).
__device__ __forceinline__ unsigned lds_off(const void* p) {
  return (unsigned)(size_t)p;
}

// CDNA5 async global->LDS copies (ASYNCcnt-tracked), per CDNA5 ISA.
__device__ __forceinline__ void async_ld128(unsigned lds, const void* g) {
  asm volatile("global_load_async_to_lds_b128 %0, %1, off" :: "v"(lds), "v"(g) : "memory");
}
__device__ __forceinline__ void async_ld64(unsigned lds, const void* g) {
  asm volatile("global_load_async_to_lds_b64 %0, %1, off" :: "v"(lds), "v"(g) : "memory");
}
__device__ __forceinline__ void wait_async0() {
  asm volatile("s_wait_asynccnt 0x0" ::: "memory");
}

// ---------------- utility kernels ----------------

__global__ void k_init_scalars(float* ss) { ss[0] = 0.f; ss[1] = 0.f; }

__global__ void k_zero(float* p, int n) {
  int i = blockIdx.x * blockDim.x + threadIdx.x;
  if (i < n) p[i] = 0.f;
}

__global__ void k_norm_sq(const float* __restrict__ x, int n, float* acc) {
  __shared__ float red[256];
  float s = 0.f;
  for (int i = blockIdx.x * blockDim.x + threadIdx.x; i < n; i += gridDim.x * blockDim.x) {
    float v = x[i];
    s += v * v;
  }
  red[threadIdx.x] = s;
  __syncthreads();
  for (int o = 128; o > 0; o >>= 1) {
    if (threadIdx.x < o) red[threadIdx.x] += red[threadIdx.x + o];
    __syncthreads();
  }
  if (threadIdx.x == 0) atomicAdd(acc, red[0]);
}

__global__ void k_finalize(const float* g1, const float* g2, float* ss) {
  ss[2] = g1[0] * rsqrtf(ss[0]);   // s1 = g1 / ||W1||
  ss[3] = g2[0] * rsqrtf(ss[1]);   // s2 = g2 / ||W2||
}

__global__ void k_cvt_bf16(const float* __restrict__ x, u16* __restrict__ y, int n) {
  for (int i = blockIdx.x * blockDim.x + threadIdx.x; i < n; i += gridDim.x * blockDim.x)
    y[i] = f2bf(x[i]);
}

// Tiled f32 -> bf16 transpose: XT[c][r] = bf16(X[r][c]). R,C multiples of 32.
// grid (C/32, R/32), block (32,8).
__global__ void k_cvt_transpose(const float* __restrict__ X, u16* __restrict__ XT,
                                int R, int C) {
  __shared__ u16 tile[32][33];
  const int tx = threadIdx.x, ty = threadIdx.y;
  const int c0 = blockIdx.x * 32, r0 = blockIdx.y * 32;
#pragma unroll
  for (int i = 0; i < 4; ++i)
    tile[ty + i * 8][tx] = f2bf(X[(size_t)(r0 + ty + i * 8) * C + (c0 + tx)]);
  __syncthreads();
#pragma unroll
  for (int i = 0; i < 4; ++i)
    XT[(size_t)(c0 + ty + i * 8) * R + (r0 + tx)] = tile[tx][ty + i * 8];
}

// ---------------- bf16 WMMA GEMM (B pre-transposed) ----------------
// C[M,N] = scale * A[M,K] x B[K,N], with B supplied transposed: BT[n][k], row
// stride ldbt. grid (N/256, M/(16*MT)), block 128 (4 waves). K%32==0.
// MT 16-row sub-tiles per block -> MT*4 WMMAs/iteration, B frags reused MT x.
// Double-buffered LDS tiles filled with async global->LDS b64/b128 copies.
template <int MT>
__global__ void __launch_bounds__(128, 1)
k_gemm_bf16t(const u16* __restrict__ A, const u16* __restrict__ BT,
             int ldbt, float* __restrict__ C, int M, int Kd, int N,
             const float* __restrict__ scale_ptr) {
  __shared__ __align__(16) u16 ldsA[2][MT * 16 * 32];   // [m][k]
  __shared__ __align__(16) u16 ldsB[2][256 * 32];       // [n][k] (transposed tile)
  const int t    = threadIdx.x;
  const int lane = t & 31;
  const int w    = t >> 5;
  const int m0   = blockIdx.y * (16 * MT);
  const int nb0  = blockIdx.x * 256;

  v8f acc[MT][4] = {};

  auto stage = [&](int buf, int k0) {
#pragma unroll
    for (int i = 0; i < 8; ++i) {
      int f0 = (t + i * 128) * 8;                 // element index in 256x32 tile
      int nl = f0 >> 5, kl = f0 & 31;
      async_ld128(lds_off(&ldsB[buf][f0]),
                  &BT[(size_t)(nb0 + nl) * ldbt + k0 + kl]);
    }
#pragma unroll
    for (int s = 0; s < MT; ++s) {
      int e0 = t * 4;                             // element index in 16x32 sub-tile
      int r = e0 >> 5, c = e0 & 31;
      async_ld64(lds_off(&ldsA[buf][s * 512 + e0]),
                 &A[(size_t)(m0 + s * 16 + r) * Kd + k0 + c]);
    }
  };

  stage(0, 0);
  const int iters = Kd >> 5;
  for (int it = 0; it < iters; ++it) {
    const int cur = it & 1;
    wait_async0();
    __syncthreads();
    if (it + 1 < iters) stage(1 - cur, (it + 1) * 32);

    Frag a[MT];
#pragma unroll
    for (int s = 0; s < MT; ++s) {
      const u16* ap = &ldsA[cur][s * 512 + (lane & 15) * 32 + (lane >> 4) * 8];
      a[s].x[0] = *(const uint4*)ap;
      a[s].x[1] = *(const uint4*)(ap + 16);
    }
#pragma unroll
    for (int f = 0; f < 4; ++f) {
      Frag bf;   // one live B fragment, reused by MT WMMAs
      const u16* bp = &ldsB[cur][(w * 64 + f * 16 + (lane & 15)) * 32 + (lane >> 4) * 8];
      bf.x[0] = *(const uint4*)bp;
      bf.x[1] = *(const uint4*)(bp + 16);
#pragma unroll
      for (int s = 0; s < MT; ++s)
        acc[s][f] = __builtin_amdgcn_wmma_f32_16x16x32_bf16(
            false, a[s].v, false, bf.v, (short)0, acc[s][f], false, false);
    }
  }

  const float scale = *scale_ptr;
#pragma unroll
  for (int s = 0; s < MT; ++s)
#pragma unroll
    for (int f = 0; f < 4; ++f) {
      int n  = nb0 + w * 64 + f * 16 + (lane & 15);
      int mb = (lane >> 4) * 8;
#pragma unroll
      for (int g = 0; g < 8; ++g)
        C[(size_t)(m0 + s * 16 + mb + g) * N + n] = scale * acc[s][f][g];
    }
}

// ---------------- fused relation kernel ----------------
// h1[p,k] = relu(pa[b,j,k] + pb[b,i,k] + pq[b,k] + b1[k]) built in LDS (bf16),
// h2 = relu(s2 * (h1 @ W2) + b2), out[b,n] += sum over pairs.
// W2 supplied transposed: W2T[n][k], row stride 1024.
// 48 pairs (3 x 16) per block: grid (4, 27, 32), block 128 (4 waves).
// 12 WMMAs per K-chunk iteration; double-buffered async W2 staging.
__global__ void __launch_bounds__(128, 1)
k_relation_fused(const float* __restrict__ pa, const float* __restrict__ pb,
                 const float* __restrict__ pq, const float* __restrict__ b1,
                 const u16* __restrict__ W2T, const float* __restrict__ b2,
                 const float* __restrict__ s2ptr, float* __restrict__ out) {
  __shared__ __align__(16) u16 ldsA[2][3 * 16 * 32];    // [p][k] relu'd h1 tiles
  __shared__ __align__(16) u16 ldsB[2][256 * 32];       // [n][k] W2 tile
  const int t    = threadIdx.x;
  const int lane = t & 31;
  const int w    = t >> 5;
  const int b    = blockIdx.z;
  const int p0   = blockIdx.y * 48;
  const int nb0  = blockIdx.x * 256;
  const float s2 = *s2ptr;

  // per-thread h1-build coordinates (4 contiguous k per thread per sub-tile)
  const int e0 = t * 4;
  const int rr = e0 >> 5, cc = e0 & 31;
  const float* paRow[3];
  const float* pbRow[3];
#pragma unroll
  for (int s = 0; s < 3; ++s) {
    int p = p0 + s * 16 + rr;
    int ii = p / 36, jj = p - ii * 36;
    paRow[s] = &pa[((size_t)b * 36 + jj) * 1024];
    pbRow[s] = &pb[((size_t)b * 36 + ii) * 1024];
  }
  const float* pqRow = &pq[(size_t)b * 1024];

  v8f acc[3][4] = {};

  auto stageB = [&](int buf, int k0) {
#pragma unroll
    for (int i = 0; i < 8; ++i) {
      int f0 = (t + i * 128) * 8;
      int nl = f0 >> 5, kl = f0 & 31;
      async_ld128(lds_off(&ldsB[buf][f0]),
                  &W2T[(size_t)(nb0 + nl) * 1024 + k0 + kl]);
    }
  };
  auto buildA = [&](int buf, int k0) {
    const int kc = k0 + cc;
    float4 Q4 = *(const float4*)&pqRow[kc];
    float4 C4 = *(const float4*)&b1[kc];
    float qx = Q4.x + C4.x, qy = Q4.y + C4.y, qz = Q4.z + C4.z, qw = Q4.w + C4.w;
#pragma unroll
    for (int s = 0; s < 3; ++s) {
      float4 A4 = *(const float4*)&paRow[s][kc];
      float4 B4 = *(const float4*)&pbRow[s][kc];
      u16 h0 = f2bf(fmaxf(A4.x + B4.x + qx, 0.f));
      u16 h1 = f2bf(fmaxf(A4.y + B4.y + qy, 0.f));
      u16 h2 = f2bf(fmaxf(A4.z + B4.z + qz, 0.f));
      u16 h3 = f2bf(fmaxf(A4.w + B4.w + qw, 0.f));
      uint2 pk;
      pk.x = (unsigned)h0 | ((unsigned)h1 << 16);
      pk.y = (unsigned)h2 | ((unsigned)h3 << 16);
      *(uint2*)&ldsA[buf][s * 512 + e0] = pk;
    }
  };

  stageB(0, 0);
  buildA(0, 0);
  for (int it = 0; it < 32; ++it) {
    const int cur = it & 1;
    wait_async0();
    __syncthreads();
    if (it + 1 < 32) {
      stageB(1 - cur, (it + 1) * 32);
      buildA(1 - cur, (it + 1) * 32);
    }

    Frag a[3];
#pragma unroll
    for (int s = 0; s < 3; ++s) {
      const u16* ap = &ldsA[cur][s * 512 + (lane & 15) * 32 + (lane >> 4) * 8];
      a[s].x[0] = *(const uint4*)ap;
      a[s].x[1] = *(const uint4*)(ap + 16);
    }
#pragma unroll
    for (int f = 0; f < 4; ++f) {
      Frag bf;   // one live B fragment, reused by 3 WMMAs
      const u16* bp = &ldsB[cur][(w * 64 + f * 16 + (lane & 15)) * 32 + (lane >> 4) * 8];
      bf.x[0] = *(const uint4*)bp;
      bf.x[1] = *(const uint4*)(bp + 16);
#pragma unroll
      for (int s = 0; s < 3; ++s)
        acc[s][f] = __builtin_amdgcn_wmma_f32_16x16x32_bf16(
            false, a[s].v, false, bf.v, (short)0, acc[s][f], false, false);
    }
  }

  // epilogue: bias + relu + row-sum over all 48 pairs + accumulate into out[b,n]
#pragma unroll
  for (int f = 0; f < 4; ++f) {
    int n = nb0 + w * 64 + f * 16 + (lane & 15);
    float bias = b2[n];
    float ssum = 0.f;
#pragma unroll
    for (int s = 0; s < 3; ++s)
#pragma unroll
      for (int g = 0; g < 8; ++g)
        ssum += fmaxf(s2 * acc[s][f][g] + bias, 0.f);
    ssum += __shfl_xor(ssum, 16, 32);     // combine M rows 0-7 and 8-15
    if (lane < 16) atomicAdd(&out[(size_t)b * 1024 + n], ssum);
  }
}

// ---------------- launch ----------------

extern "C" void kernel_launch(void* const* d_in, const int* in_sizes, int n_in,
                              void* d_out, int out_size, void* d_ws, size_t ws_size,
                              hipStream_t stream) {
  (void)in_sizes; (void)n_in; (void)out_size; (void)ws_size;
  const float* v  = (const float*)d_in[0];   // (32,36,2048)
  const float* q  = (const float*)d_in[1];   // (32,1024)
  const float* W1 = (const float*)d_in[2];   // (5120,1024)
  const float* g1 = (const float*)d_in[3];
  const float* b1 = (const float*)d_in[4];   // (1024,)
  const float* W2 = (const float*)d_in[5];   // (1024,1024)
  const float* g2 = (const float*)d_in[6];
  const float* b2 = (const float*)d_in[7];   // (1024,)
  float* out = (float*)d_out;                // (32,1024)

  char* wsp = (char*)d_ws;
  float* ss = (float*)wsp;                   // [ss1, ss2, s1, s2]
  size_t off = 256;
  u16* v_bf = (u16*)(wsp + off); off += (size_t)32 * 36 * 2048 * 2;
  u16* q_bf = (u16*)(wsp + off); off += (size_t)32 * 1024 * 2;
  u16* W1T  = (u16*)(wsp + off); off += (size_t)5120 * 1024 * 2;  // [1024][5120]
  u16* W2T  = (u16*)(wsp + off); off += (size_t)1024 * 1024 * 2;  // [1024][1024]
  float* pa = (float*)(wsp + off); off += (size_t)32 * 36 * 1024 * 4;
  float* pb = (float*)(wsp + off); off += (size_t)32 * 36 * 1024 * 4;
  float* pq = (float*)(wsp + off);

  k_init_scalars<<<1, 1, 0, stream>>>(ss);
  k_zero<<<128, 256, 0, stream>>>(out, 32 * 1024);
  k_norm_sq<<<512, 256, 0, stream>>>(W1, 5120 * 1024, ss + 0);
  k_norm_sq<<<512, 256, 0, stream>>>(W2, 1024 * 1024, ss + 1);
  k_finalize<<<1, 1, 0, stream>>>(g1, g2, ss);

  k_cvt_bf16<<<2048, 256, 0, stream>>>(v, v_bf, 32 * 36 * 2048);
  k_cvt_bf16<<<128,  256, 0, stream>>>(q, q_bf, 32 * 1024);
  k_cvt_transpose<<<dim3(32, 160), dim3(32, 8), 0, stream>>>(W1, W1T, 5120, 1024);
  k_cvt_transpose<<<dim3(32, 32),  dim3(32, 8), 0, stream>>>(W2, W2T, 1024, 1024);

  // pa = s1 * v @ W1[:2048], pb = s1 * v @ W1[2048:4096], pq = s1 * q @ W1[4096:]
  k_gemm_bf16t<3><<<dim3(4, 24), 128, 0, stream>>>(v_bf, W1T,        5120, pa, 1152, 2048, 1024, ss + 2);
  k_gemm_bf16t<3><<<dim3(4, 24), 128, 0, stream>>>(v_bf, W1T + 2048, 5120, pb, 1152, 2048, 1024, ss + 2);
  k_gemm_bf16t<2><<<dim3(4, 1),  128, 0, stream>>>(q_bf, W1T + 4096, 5120, pq,   32, 1024, 1024, ss + 2);

  // fused h1-build + h2 GEMM + reduction (87 GFLOP path)
  k_relation_fused<<<dim3(4, 27, 32), 128, 0, stream>>>(pa, pb, pq, b1, W2T, b2, ss + 3, out);
}
